// MILossGaussian_21397527069320
// MI455X (gfx1250) — matrix operand order repaired and checked by reading, not compiled, not run on hardware
//
#include <hip/hip_runtime.h>

// ---------------------------------------------------------------------------
// MI loss with Gaussian KDE soft histograms, CDNA5 (gfx1250, wave32, WMMA)
// ---------------------------------------------------------------------------

typedef _Float16 h4_t  __attribute__((ext_vector_type(4)));
typedef _Float16 h8_t  __attribute__((ext_vector_type(8)));
typedef _Float16 h16_t __attribute__((ext_vector_type(16)));
typedef float    f4_t  __attribute__((ext_vector_type(4)));
typedef float    f8_t  __attribute__((ext_vector_type(8)));

#define NBINS   64
#define CHUNK   128
#define PADW    136                 // 128 + 8 f16 pad -> 272B row stride (bank-shift 4/row)
#define PVOX    (128*128*128)       // voxels per batch (2,097,152)
#define NCHUNKS (PVOX / CHUNK)      // 16384
#define NWG_X   512

// KDE constants (constant-folded)
constexpr double kBinW       = 1.0 / 64.0;
constexpr double kSigma      = kBinW / 2.3548200450309493;          // FWHM -> sigma
constexpr double kInvTwoSig2 = 1.0 / (2.0 * kSigma * kSigma);
constexpr double kScaleLog2  = kInvTwoSig2 * 1.4426950408889634;    // *log2(e) for exp2
constexpr double kGaussNorm  = 1.0 / (2.5066282746310002 * kSigma); // 1/(sqrt(2pi)*sigma)
constexpr double kBinStep    = 1.0 / 63.0;                          // linspace(0,1,64) spacing
#define EPS 1e-5f

__device__ __forceinline__ h16_t cat8(h8_t a, h8_t b) {
  return __builtin_shufflevector(a, b, 0,1,2,3,4,5,6,7,8,9,10,11,12,13,14,15);
}

// raw v_exp_f32: args here are in [-16384, 0]; hardware underflow-to-zero is
// exactly the desired far-bin behavior, so skip the OCML range-guard fixup.
__device__ __forceinline__ float fast_exp2(float a) {
  return __builtin_amdgcn_exp2f(a);
}
__device__ __forceinline__ float fast_log(float a) {
  return __builtin_amdgcn_logf(a) * 0.69314718055994531f;  // ln(x) = log2(x)*ln2
}

// ---------------------------------------------------------------------------
// Stage 0: zero the histogram accumulator in workspace
// ---------------------------------------------------------------------------
__global__ void mi_zero_kernel(float* __restrict__ p, int n) {
  int i = blockIdx.x * blockDim.x + threadIdx.x;
  if (i < n) p[i] = 0.0f;
}

// ---------------------------------------------------------------------------
// Stage 1: fused KDE-window + WMMA outer-product histogram
//   grid  = (NWG_X, 2)  [x: chunk stripes, y: batch]
//   block = 256 threads = 8 waves; 64x64 joint hist = 4x4 grid of 16x16 tiles,
//   wave w owns tiles (bi = w>>1, cj = 2*(w&1) and 2*(w&1)+1).
//   No divergence anywhere: EXEC is all-ones for every WMMA.
// ---------------------------------------------------------------------------
__global__ __launch_bounds__(256)
void mi_hist_kernel(const float* __restrict__ x,
                    const float* __restrict__ y,
                    float* __restrict__ hist /* [2][64][64] */) {
  __shared__ __align__(16) _Float16 sWx[NBINS][PADW];
  __shared__ __align__(16) _Float16 sWy[NBINS][PADW];

  const int t    = threadIdx.x;
  const int lane = t & 31;
  const int wave = t >> 5;
  const int n    = blockIdx.y;

  // window-phase role: 4-voxel group vg, bins wave + 8*j
  const int vg = (t & 31) * 4;          // voxels vg..vg+3
  const int b0 = wave;                  // bins b0, b0+8, ..., b0+56

  // compute-phase role
  const int bi   = wave >> 1;           // A-tile row   (0..3)
  const int cj0  = (wave & 1) * 2;      // B-tile cols  {cj0, cj0+1}
  const int rowA = bi * 16 + (lane & 15);
  const int rowB = cj0 * 16 + (lane & 15);
  const int koffA = (lane >= 16) ? 8 : 0;   // A frag: K runs {koffA..+7},{koffA+16..+23}
  const int koffB = (lane >= 16) ? 8 : 0;   // B frag: K runs {koffB..+7},{koffB+16..+23}

  const float sc = (float)kScaleLog2;
  const float gn = (float)kGaussNorm;

  f8_t acc0 = {};
  f8_t acc1 = {};

  const long long base_n = (long long)n * PVOX;

  // software pipeline: preload this WG's first chunk (global_load_b128)
  const long long base0 = base_n + (long long)blockIdx.x * CHUNK;
  f4_t xv = *(const f4_t*)&x[base0 + vg];
  f4_t yv = *(const f4_t*)&y[base0 + vg];

  for (int chunk = blockIdx.x; chunk < NCHUNKS; chunk += NWG_X) {
    // --- KDE windows: 8192 (bin,voxel) values, each computed exactly once ---
    // thread -> (bin b0+8j, voxels vg..vg+3); wave writes one full 256B row
#pragma unroll
    for (int j = 0; j < 8; ++j) {
      const int b = b0 + 8 * j;
      const float c = (float)b * (float)kBinStep;
      h4_t hx, hy;
#pragma unroll
      for (int i = 0; i < 4; ++i) {
        const float dx = xv[i] - c;
        const float dy = yv[i] - c;
        hx[i] = (_Float16)(fast_exp2(-(dx * dx) * sc) * gn);
        hy[i] = (_Float16)(fast_exp2(-(dy * dy) * sc) * gn);
      }
      *(h4_t*)&sWx[b][vg] = hx;   // ds_store_b64, conflict-free
      *(h4_t*)&sWy[b][vg] = hy;
    }
    __syncthreads();

    // issue next chunk's loads now; latency hides behind the WMMA phase
    const int nxt = chunk + NWG_X;
    if (nxt < NCHUNKS) {
      const long long nb = base_n + (long long)nxt * CHUNK;
      xv = *(const f4_t*)&x[nb + vg];
      yv = *(const f4_t*)&y[nb + vg];
    }

    // --- 16x16x32 WMMA rank-updates over the chunk ---
#pragma unroll
    for (int kk = 0; kk < CHUNK; kk += 32) {
      h8_t a0 = *(const h8_t*)&sWx[rowA][kk + koffA];
      h8_t a1 = *(const h8_t*)&sWx[rowA][kk + koffA + 16];
      h16_t A = cat8(a0, a1);

      h8_t p0 = *(const h8_t*)&sWy[rowB][kk + koffB];
      h8_t p1 = *(const h8_t*)&sWy[rowB][kk + koffB + 16];
      h16_t B0 = cat8(p0, p1);

      h8_t q0 = *(const h8_t*)&sWy[rowB + 16][kk + koffB];
      h8_t q1 = *(const h8_t*)&sWy[rowB + 16][kk + koffB + 16];
      h16_t B1 = cat8(q0, q1);

      acc0 = __builtin_amdgcn_wmma_f32_16x16x32_f16(
          false, A, false, B0, (short)0, acc0, false, false);
      acc1 = __builtin_amdgcn_wmma_f32_16x16x32_f16(
          false, A, false, B1, (short)0, acc1, false, false);
    }
    __syncthreads();   // before next chunk overwrites sWx/sWy
  }

  // --- flush accumulated tiles: D layout (VGPR r, lane) -> (M, N) ---
  const int Mb = (lane >= 16) ? 8 : 0;
  const int N  = lane & 15;
  float* hn = hist + n * NBINS * NBINS;
#pragma unroll
  for (int r = 0; r < 8; ++r) {
    const int M = r + Mb;
    atomicAdd(&hn[(bi * 16 + M) * NBINS + (cj0 * 16 + N)],       acc0[r]);
    atomicAdd(&hn[(bi * 16 + M) * NBINS + ((cj0 + 1) * 16 + N)], acc1[r]);
  }
}

// ---------------------------------------------------------------------------
// Stage 2: entropies + loss (single block; 8K elements -> negligible time)
// ---------------------------------------------------------------------------
__device__ __forceinline__ float block_reduce256(float v, float* red, int t) {
  red[t] = v;
  __syncthreads();
  for (int o = 128; o > 0; o >>= 1) {
    if (t < o) red[t] += red[t + o];
    __syncthreads();
  }
  float r = red[0];
  __syncthreads();
  return r;
}

__global__ __launch_bounds__(256)
void mi_finalize_kernel(const float* __restrict__ hist, float* __restrict__ out) {
  __shared__ float red[256];
  const int t = threadIdx.x;

  float loss = 0.0f;
  for (int n = 0; n < 2; ++n) {
    const float* h = hist + n * NBINS * NBINS;

    // total mass
    float s = 0.0f;
    for (int i = t; i < NBINS * NBINS; i += 256) s += h[i];
    const float norm = block_reduce256(s, red, t) + EPS;

    // joint entropy
    float ej = 0.0f;
    for (int i = t; i < NBINS * NBINS; i += 256) {
      const float p = h[i] / norm;
      ej += p * fast_log(p + EPS);
    }
    const float entJ = -block_reduce256(ej, red, t);

    // marginal entropies (row sums -> p_x, col sums -> p_y)
    float ex = 0.0f, ey = 0.0f;
    if (t < NBINS) {
      float px = 0.0f, py = 0.0f;
      for (int c = 0; c < NBINS; ++c) {
        px += h[t * NBINS + c];
        py += h[c * NBINS + t];
      }
      px /= norm;
      py /= norm;
      ex = px * fast_log(px + EPS);
      ey = py * fast_log(py + EPS);
    }
    const float entX = -block_reduce256(ex, red, t);
    const float entY = -block_reduce256(ey, red, t);

    loss += (entX + entY) / entJ;
  }

  if (t == 0) out[0] = -loss * 0.5f;   // -mean over N=2
}

// ---------------------------------------------------------------------------
extern "C" void kernel_launch(void* const* d_in, const int* in_sizes, int n_in,
                              void* d_out, int out_size, void* d_ws, size_t ws_size,
                              hipStream_t stream) {
  (void)in_sizes; (void)n_in; (void)out_size; (void)ws_size;

  const float* x = (const float*)d_in[0];
  const float* y = (const float*)d_in[1];
  float* hist = (float*)d_ws;            // 2 * 64 * 64 floats = 32 KB
  float* out  = (float*)d_out;

  const int histN = 2 * NBINS * NBINS;
  mi_zero_kernel<<<(histN + 255) / 256, 256, 0, stream>>>(hist, histN);

  dim3 grid(NWG_X, 2);
  mi_hist_kernel<<<grid, 256, 0, stream>>>(x, y, hist);

  mi_finalize_kernel<<<1, 256, 0, stream>>>(hist, out);
}
